// LatentDistribution_54434415509958
// MI455X (gfx1250) — matrix operand order repaired and checked by reading, not compiled, not run on hardware
//
#include <hip/hip_runtime.h>
#include <hip/hip_bf16.h>

// ---------------------------------------------------------------------------
// LatentDistribution: mu = relu(z @ W_mu^T + b_mu); ls = relu(z @ W_std^T + b_std)
// kl = mean_over_rows( 0.5 * sum_n[ (1+mu^2)*exp(-2 ls) - 1 + 2 ls ] )
// out = [ mu (BS*256 f32) | kl (1 f32) ]
// BS = 131072, DIM = 256.
// Strategy: bf16 WMMA (v_wmma_f32_16x16x32_bf16), f32 accumulate.
// Register blocking: each wave owns 32 rows (two 16x16 A tiles), so every
// B fragment load feeds 4 WMMAs (2 row tiles x {mu, std}) -> halves L2 traffic.
// ---------------------------------------------------------------------------

typedef __attribute__((ext_vector_type(16))) __bf16 v16bf;
typedef __attribute__((ext_vector_type(8)))  __bf16 v8bf;
typedef __attribute__((ext_vector_type(4)))  __bf16 v4bf;
typedef __attribute__((ext_vector_type(8)))  float  v8f;

#define LD_DIM   256
#define LD_BS    131072
#define LD_MTILE 256          // rows per block (8 waves * 32 rows each)

// ---------------------------------------------------------------------------
// Kernel 0: convert W_mu and W_std (f32 row-major 256x256) to bf16 in d_ws,
// and zero the KL accumulator slot in d_out.
// ---------------------------------------------------------------------------
__global__ __launch_bounds__(256) void ld_convert_weights(
    const float* __restrict__ Wmu, const float* __restrict__ Wstd,
    __bf16* __restrict__ wbf, float* __restrict__ kl_slot)
{
    const int idx = blockIdx.x * 256 + threadIdx.x;   // 32768 threads, 4 elems each
    const int N = LD_DIM * LD_DIM;
    int e = idx * 4;
    const float* src = (e < N) ? (Wmu + e) : (Wstd + (e - N));
    float4 v = *(const float4*)src;
    v4bf o;
    o[0] = (__bf16)v.x; o[1] = (__bf16)v.y; o[2] = (__bf16)v.z; o[3] = (__bf16)v.w;
    *(v4bf*)(wbf + e) = o;
    if (idx == 0) *kl_slot = 0.0f;
}

// ---------------------------------------------------------------------------
// Kernel 1: fused GEMMs + epilogue.
//   grid = BS/256 blocks, 256 threads (8 wave32).
//   Wave w owns rows [block*256 + w*32, +32) as two 16-row A tiles.
// ---------------------------------------------------------------------------
__global__ __launch_bounds__(256) void ld_main_kernel(
    const float*  __restrict__ z,
    const __bf16* __restrict__ wbf,
    const float*  __restrict__ bmu,
    const float*  __restrict__ bsd,
    float*        __restrict__ out,
    float*        __restrict__ kl_slot)
{
    __shared__ __bf16 zs[LD_MTILE * LD_DIM];   // 128 KB staged z tile (bf16)
    __shared__ float  red[256];

    const int tid   = threadIdx.x;
    const int lane  = tid & 31;
    const int wave  = tid >> 5;
    const int mrow  = lane & 15;       // row within a 16-row tile (A frag)
    const int khalf = lane >> 4;       // which K half-chunk this lane holds
    const long base_row = (long)blockIdx.x * LD_MTILE;

    // ---- Stage z tile (f32 global -> bf16 LDS), fully coalesced float4 loads.
    {
        const float4* zg = (const float4*)(z + base_row * LD_DIM);
        #pragma unroll
        for (int i = 0; i < (LD_MTILE * LD_DIM / 4) / 256; ++i) {   // 64 iters
            int j = tid + i * 256;
            float4 v = zg[j];
            v4bf o;
            o[0] = (__bf16)v.x; o[1] = (__bf16)v.y;
            o[2] = (__bf16)v.z; o[3] = (__bf16)v.w;
            *(v4bf*)(&zs[4 * j]) = o;
        }
    }
    __syncthreads();

    // ---- Preload the 16 A fragments (two 16-row tiles x 8 K-steps) from LDS.
    // A frag, contiguous k order (B uses the same): lane<16 holds row=lane, K = kt*32+0..15;
    // lane>=16 holds row=lane-16, K = kt*32+16..31.
    v16bf afrag0[8];
    v16bf afrag1[8];
    {
        const __bf16* zrow0 = &zs[(wave * 32 + mrow) * LD_DIM];
        const __bf16* zrow1 = zrow0 + 16 * LD_DIM;
        #pragma unroll
        for (int kt = 0; kt < 8; ++kt) {
            v8bf lo0 = *(const v8bf*)(zrow0 + kt * 32 + khalf * 16);
            v8bf hi0 = *(const v8bf*)(zrow0 + kt * 32 + khalf * 16 + 8);
            v8bf lo1 = *(const v8bf*)(zrow1 + kt * 32 + khalf * 16);
            v8bf hi1 = *(const v8bf*)(zrow1 + kt * 32 + khalf * 16 + 8);
            v16bf a0, a1;
            #pragma unroll
            for (int e = 0; e < 8; ++e) {
                a0[e] = lo0[e]; a0[e + 8] = hi0[e];
                a1[e] = lo1[e]; a1[e + 8] = hi1[e];
            }
            afrag0[kt] = a0;
            afrag1[kt] = a1;
        }
    }

    const __bf16* wmu = wbf;
    const __bf16* wsd = wbf + LD_DIM * LD_DIM;

    float kl_acc = 0.0f;

    for (int nt = 0; nt < 16; ++nt) {
        v8f accM0 = {}, accM1 = {};
        v8f accS0 = {}, accS1 = {};

        // B frag, contiguous k order matching A: lane n (0..15) holds column n, K=0..15;
        // lane n+16 holds column n, K=16..31. W row-major => contiguous 32B.
        const int bcol = nt * 16 + mrow;                    // output feature index
        const __bf16* wmrow = wmu + bcol * LD_DIM + khalf * 16;
        const __bf16* wsrow = wsd + bcol * LD_DIM + khalf * 16;

        #pragma unroll
        for (int kt = 0; kt < 8; ++kt) {
            v16bf bm = *(const v16bf*)(wmrow + kt * 32);
            v16bf bs = *(const v16bf*)(wsrow + kt * 32);
            accM0 = __builtin_amdgcn_wmma_f32_16x16x32_bf16(
                        false, afrag0[kt], false, bm, (short)0, accM0, false, false);
            accM1 = __builtin_amdgcn_wmma_f32_16x16x32_bf16(
                        false, afrag1[kt], false, bm, (short)0, accM1, false, false);
            accS0 = __builtin_amdgcn_wmma_f32_16x16x32_bf16(
                        false, afrag0[kt], false, bs, (short)0, accS0, false, false);
            accS1 = __builtin_amdgcn_wmma_f32_16x16x32_bf16(
                        false, afrag1[kt], false, bs, (short)0, accS1, false, false);
        }

        // ---- Epilogue for the two 16x16 tiles (all in-register per lane).
        // C/D layout: VGPR v, lanes 0-15 -> (M=v, N=lane); lanes 16-31 -> (M=v+8).
        const float bm_n = bmu[bcol];
        const float bs_n = bsd[bcol];
        const long  rowbase0 = base_row + wave * 32 + 8 * khalf;
        float* orow0 = out + rowbase0 * LD_DIM + bcol;
        float* orow1 = orow0 + 16L * LD_DIM;

        #pragma unroll
        for (int v = 0; v < 8; ++v) {
            float mu0 = accM0[v] + bm_n; mu0 = mu0 > 0.0f ? mu0 : 0.0f;
            float ls0 = accS0[v] + bs_n; ls0 = ls0 > 0.0f ? ls0 : 0.0f;
            float iv0 = __expf(-2.0f * ls0);
            kl_acc += (1.0f + mu0 * mu0) * iv0 - 1.0f + 2.0f * ls0;
            orow0[(long)v * LD_DIM] = mu0;

            float mu1 = accM1[v] + bm_n; mu1 = mu1 > 0.0f ? mu1 : 0.0f;
            float ls1 = accS1[v] + bs_n; ls1 = ls1 > 0.0f ? ls1 : 0.0f;
            float iv1 = __expf(-2.0f * ls1);
            kl_acc += (1.0f + mu1 * mu1) * iv1 - 1.0f + 2.0f * ls1;
            orow1[(long)v * LD_DIM] = mu1;
        }
    }

    // ---- Block-reduce KL partials, one fp32 atomic per block.
    red[tid] = kl_acc * (0.5f / (float)LD_BS);
    __syncthreads();
    #pragma unroll
    for (int s = 128; s > 0; s >>= 1) {
        if (tid < s) red[tid] += red[tid + s];
        __syncthreads();
    }
    if (tid == 0) {
        __hip_atomic_fetch_add(kl_slot, red[0], __ATOMIC_RELAXED,
                               __HIP_MEMORY_SCOPE_AGENT);
    }
}

// ---------------------------------------------------------------------------
// Launch
// ---------------------------------------------------------------------------
extern "C" void kernel_launch(void* const* d_in, const int* in_sizes, int n_in,
                              void* d_out, int out_size, void* d_ws, size_t ws_size,
                              hipStream_t stream)
{
    const float* z    = (const float*)d_in[0];
    const float* W_mu = (const float*)d_in[1];
    const float* b_mu = (const float*)d_in[2];
    const float* W_st = (const float*)d_in[3];
    const float* b_st = (const float*)d_in[4];

    float*  out     = (float*)d_out;
    float*  kl_slot = out + (long)LD_BS * LD_DIM;
    __bf16* wbf     = (__bf16*)d_ws;       // needs 256 KB

    // Kernel 0: weight conversion (131072 elems / 4 per thread / 256 per block)
    ld_convert_weights<<<128, 256, 0, stream>>>(W_mu, W_st, wbf, kl_slot);

    // Kernel 1: fused GEMMs + epilogue
    ld_main_kernel<<<LD_BS / LD_MTILE, 256, 0, stream>>>(z, wbf, b_mu, b_st,
                                                         out, kl_slot);
}
